// AttentionLayer_34797825032519
// MI455X (gfx1250) — matrix operand order repaired and checked by reading, not compiled, not run on hardware
//
#include <hip/hip_runtime.h>
#include <math.h>
#include <stddef.h>
#include <stdint.h>

// ---------------- Types for CDNA5 WMMA / TDM ----------------
typedef __attribute__((ext_vector_type(16))) __bf16 v16bf;
typedef __attribute__((ext_vector_type(8)))  float  v8f;
typedef __attribute__((ext_vector_type(4)))  unsigned int v4u;
typedef __attribute__((ext_vector_type(8)))  int v8i;
typedef __attribute__((ext_vector_type(4)))  int v4i;
typedef unsigned short bf16s;   // bf16 storage (bit pattern)

#define BB 2
#define LL 2048
#define SSZ 2048
#define DM 1024
#define HH 16
#define DH 64

#define SC_FLOATS (16 * SSZ)        // 16x2048 fp32 score row: 128 KB
#define KCHUNK 128                  // keys staged per TDM op
#define STAGE_BYTES (KCHUNK * DH * 2)  // 16 KB per staging buffer

// ---------------- bf16 helpers (round-to-nearest-even) ----------------
__device__ __forceinline__ unsigned short f2bf(float f) {
  unsigned u = __float_as_uint(f);
  u = (u + 0x7FFFu + ((u >> 16) & 1u)) >> 16;
  return (unsigned short)u;
}
__device__ __forceinline__ unsigned pack2bf(float lo, float hi) {
  return (unsigned)f2bf(lo) | ((unsigned)f2bf(hi) << 16);
}

// ---------------- Fragment loaders ----------------
// A-matrix 16x32 bf16 (ISA layout): lane&15 = row M; ks=(lane>>4)*8;
// elements 0..7 = K[ks..ks+7], elements 8..15 = K[ks+16..ks+23].
__device__ __forceinline__ v16bf load_a_bf16(const bf16s* base, int ld, int lane) {
  const int row = lane & 15;
  const int ks  = (lane >> 4) * 8;
  const bf16s* p0 = base + row * ld + ks;
  union { v16bf v; uint4 q[2]; } r;
  r.q[0] = *(const uint4*)(p0);
  r.q[1] = *(const uint4*)(p0 + 16);
  return r.v;
}

// Same A layout but source is fp32 (global or LDS via generic ptr), converted on the fly.
__device__ __forceinline__ v16bf load_a_f32(const float* base, int ld, int lane) {
  const int row = lane & 15;
  const int ks  = (lane >> 4) * 8;
  const float* p0 = base + row * ld + ks;
  const float* p1 = p0 + 16;
  float4 a0 = *(const float4*)(p0);
  float4 a1 = *(const float4*)(p0 + 4);
  float4 b0 = *(const float4*)(p1);
  float4 b1 = *(const float4*)(p1 + 4);
  union { v16bf v; unsigned u[8]; } r;
  r.u[0] = pack2bf(a0.x, a0.y);
  r.u[1] = pack2bf(a0.z, a0.w);
  r.u[2] = pack2bf(a1.x, a1.y);
  r.u[3] = pack2bf(a1.z, a1.w);
  r.u[4] = pack2bf(b0.x, b0.y);
  r.u[5] = pack2bf(b0.z, b0.w);
  r.u[6] = pack2bf(b1.x, b1.y);
  r.u[7] = pack2bf(b1.z, b1.w);
  return r.v;
}

// B-matrix 32x16 bf16: lane&15 = column N; ks=(lane>>4)*8; elements 0..7 = K[ks..ks+7],
// elements 8..15 = K[ks+16..ks+23]. Source stored K-contiguous per column.
__device__ __forceinline__ v16bf load_b_bf16(const bf16s* base, int ld, int lane) {
  const int n  = lane & 15;
  const int ks = (lane >> 4) * 8;
  const bf16s* p = base + n * ld + ks;
  union { v16bf v; uint4 q[2]; } r;
  r.q[0] = *(const uint4*)(p);
  r.q[1] = *(const uint4*)(p + 16);
  return r.v;
}

__device__ __forceinline__ v8f wmma_bf16(v16bf a, v16bf b, v8f c) {
  return __builtin_amdgcn_wmma_f32_16x16x32_bf16(false, a, false, b, (short)0, c,
                                                 false, false);
}

__device__ __forceinline__ v8f zero_v8f() {
  v8f z = {0.f, 0.f, 0.f, 0.f, 0.f, 0.f, 0.f, 0.f};
  return z;
}

// ---------------- TDM: 1-D contiguous bulk copy global -> LDS ----------------
// Copies `bytes` (multiple of 8) from gaddr to LDS byte offset lds_off.
// D# per CDNA5 ISA 8.3/8.4: group0={count=1, lds_addr, global_addr, type=2},
// group1={data_size=8B, tensor_dim0=units, tensor_dim1=1, tile_dim0=units}.
__device__ __forceinline__ void tdm_load_1d(unsigned long long gaddr,
                                            unsigned lds_off, unsigned bytes) {
  const unsigned units = bytes >> 3;  // 8-byte elements
  v4u g0;
  g0[0] = 1u;                                         // count=1 (valid)
  g0[1] = lds_off;                                    // lds_addr
  g0[2] = (unsigned)(gaddr & 0xFFFFFFFFull);          // global_addr[31:0]
  g0[3] = (unsigned)((gaddr >> 32) & 0x01FFFFFFull)   // global_addr[56:32]
          | (2u << 30);                               // type=2 ("image")
  v8i g1;
  g1[0] = 0x00030000;                 // data_size=3 (8B); mask/flags = 0
  g1[1] = (int)((units & 0xFFFFu) << 16);   // tensor_dim0[15:0]
  g1[2] = (int)((units >> 16) | (1u << 16)); // tensor_dim0[31:16], tensor_dim1=1
  g1[3] = (int)((units & 0xFFFFu) << 16);   // tile_dim0 = units (<=65535)
  g1[4] = 0;                          // tile_dim1=0 (1-D), tile_dim2=0
  g1[5] = (int)units;                 // tensor_dim0_stride
  g1[6] = 0;
  g1[7] = 0;
  v4i gz = {0, 0, 0, 0};
#if defined(__clang_major__) && (__clang_major__ >= 23)
  v8i gz8 = {0, 0, 0, 0, 0, 0, 0, 0};
  __builtin_amdgcn_tensor_load_to_lds(g0, g1, gz, gz, gz8, 0);
#else
  __builtin_amdgcn_tensor_load_to_lds(g0, g1, gz, gz, 0);
#endif
}

// ---------------- Kernel 0: weight transpose fp32[K,N] -> bf16[N,K] ----------------
__global__ __launch_bounds__(256) void wtrans_kernel(const float* __restrict__ W,
                                                     bf16s* __restrict__ Wt) {
  const int idx = blockIdx.x * 256 + threadIdx.x;  // 0 .. 1024*1024-1
  const int k = idx >> 10;
  const int n = idx & 1023;
  Wt[n * DM + k] = f2bf(W[k * DM + n]);
}

// ---------------- Kernel 1/3: projection GEMM ----------------
// C[M=4096, N=1024] = A[M,1024] @ W[1024,N] + bias
// MODE 0: fp32 out row-major          (final projection -> d_out)
// MODE 1: bf16 out [b,h,l,d]          (Q and K buffers)
// MODE 2: bf16 out [b,h,d,s]          (V buffer, transposed)
template <int MODE>
__global__ __launch_bounds__(256) void gemm_proj(const float* __restrict__ A,
                                                 const bf16s* __restrict__ Bt,
                                                 const float* __restrict__ bias,
                                                 float* __restrict__ Cf,
                                                 bf16s* __restrict__ Cb) {
  const int lane  = threadIdx.x & 31;
  const int w     = threadIdx.x >> 5;
  const int waveM = w & 3;
  const int waveN = w >> 2;
  const int row0  = blockIdx.x * 64 + waveM * 16;
  const int col0  = blockIdx.y * 128 + waveN * 64;

  v8f acc[4];
#pragma unroll
  for (int s = 0; s < 4; ++s) acc[s] = zero_v8f();

  for (int kk = 0; kk < DM; kk += 32) {
    if (kk + 32 < DM)  // prefetch next A K-slice (global_prefetch_b8)
      __builtin_prefetch(A + (size_t)(row0 + (lane & 15)) * DM + kk + 32, 0, 3);
    v16bf af = load_a_f32(A + (size_t)row0 * DM + kk, DM, lane);
#pragma unroll
    for (int s = 0; s < 4; ++s) {
      v16bf bf = load_b_bf16(Bt + (size_t)(col0 + s * 16) * DM + kk, DM, lane);
      acc[s] = wmma_bf16(af, bf, acc[s]);
    }
  }

  const int rsel = (lane < 16) ? 0 : 8;
#pragma unroll
  for (int s = 0; s < 4; ++s) {
    const int n = col0 + s * 16 + (lane & 15);
    const float bv = bias[n];
#pragma unroll
    for (int i = 0; i < 8; ++i) {
      const int r = row0 + rsel + i;
      const float val = acc[s][i] + bv;
      if (MODE == 0) {
        Cf[(size_t)r * DM + n] = val;
      } else if (MODE == 1) {
        const int b_ = r >> 11;   // / 2048
        const int l  = r & 2047;
        const int h  = n >> 6;
        const int d  = n & 63;
        Cb[(((size_t)(b_ * HH + h) * LL) + l) * DH + d] = f2bf(val);
      } else {
        const int b_ = r >> 11;
        const int sidx = r & 2047;
        const int h  = n >> 6;
        const int d  = n & 63;
        Cb[(((size_t)(b_ * HH + h) * DH) + d) * SSZ + sidx] = f2bf(val);
      }
    }
  }
}

// ---------------- Kernel 2: fused attention ----------------
// One wave (32 threads) per (b, h, 16-query tile). Full 16x2048 fp32 score
// row in LDS (128 KB) + 2x16 KB TDM-staged K chunks => 160 KB/WG, exactly
// two workgroups per 320 KB WGP. K is streamed via tensor_load_to_lds with
// double buffering (TENSORcnt), overlapping DMA with WMMA.
__global__ __launch_bounds__(32) void attn_fused(const bf16s* __restrict__ Qb,
                                                 const bf16s* __restrict__ Kb,
                                                 const bf16s* __restrict__ Vt,
                                                 float* __restrict__ Aout,
                                                 float* __restrict__ Of) {
  extern __shared__ float sc[];  // [16 x 2048 fp32 scores][2 x 16KB K stage]
  bf16s* kstage = (bf16s*)(sc + SC_FLOATS);

  const int lane = threadIdx.x;
  const int bh = blockIdx.x >> 7;  // b*H + h
  const int qt = blockIdx.x & 127;
  const int q0 = qt * 16;
  const int b_ = bh >> 4;
  const int h  = bh & 15;
  const float scale = 0.125f;  // 1/sqrt(64)

  // dynamic LDS starts right after the static group segment
  const unsigned lds_stage0 =
      __builtin_amdgcn_groupstaticsize() + (unsigned)(SC_FLOATS * 4);

  const bf16s* qbase = Qb + ((size_t)bh * LL + q0) * DH;
  const v16bf aq0 = load_a_bf16(qbase, DH, lane);
  const v16bf aq1 = load_a_bf16(qbase + 32, DH, lane);

  const int colL = lane & 15;
  const int rsel = (lane < 16) ? 0 : 8;

  float m8[8], s8[8];
#pragma unroll
  for (int i = 0; i < 8; ++i) { m8[i] = -3.0e38f; s8[i] = 0.f; }

  const unsigned long long kghead =
      (unsigned long long)(uintptr_t)(Kb + (size_t)bh * SSZ * DH);

  // Pass 1: scores -> LDS, streaming row-max; K staged by TDM, double-buffered
  tdm_load_1d(kghead, lds_stage0, STAGE_BYTES);
  for (int c = 0; c < SSZ / KCHUNK; ++c) {
    if (c + 1 < SSZ / KCHUNK) {
      tdm_load_1d(kghead + (unsigned long long)(c + 1) * STAGE_BYTES,
                  lds_stage0 + (unsigned)((c + 1) & 1) * STAGE_BYTES,
                  STAGE_BYTES);
      __builtin_amdgcn_s_wait_tensorcnt(1);  // chunk c landed
    } else {
      __builtin_amdgcn_s_wait_tensorcnt(0);
    }
    const bf16s* kc = kstage + (c & 1) * (STAGE_BYTES / 2);
#pragma unroll 1
    for (int t2 = 0; t2 < KCHUNK / 16; ++t2) {
      const int t = c * (KCHUNK / 16) + t2;
      v16bf bk0 = load_b_bf16(kc + t2 * 16 * DH, DH, lane);
      v16bf bk1 = load_b_bf16(kc + t2 * 16 * DH + 32, DH, lane);
      v8f cc = zero_v8f();
      cc = wmma_bf16(aq0, bk0, cc);
      cc = wmma_bf16(aq1, bk1, cc);
#pragma unroll
      for (int i = 0; i < 8; ++i) {
        const float sv = cc[i] * scale;
        sc[(rsel + i) * SSZ + t * 16 + colL] = sv;
        m8[i] = fmaxf(m8[i], sv);
      }
    }
  }
  // reduce max within each 16-lane half (rows 0..7 / 8..15)
#pragma unroll
  for (int i = 0; i < 8; ++i) {
    float m = m8[i];
    m = fmaxf(m, __shfl_xor(m, 1, 16));
    m = fmaxf(m, __shfl_xor(m, 2, 16));
    m = fmaxf(m, __shfl_xor(m, 4, 16));
    m = fmaxf(m, __shfl_xor(m, 8, 16));
    m8[i] = m;
  }

  // Pass 2: exp + row-sum (overwrite LDS with exp values)
  for (int t = 0; t < SSZ / 16; ++t) {
#pragma unroll
    for (int i = 0; i < 8; ++i) {
      const int idx = (rsel + i) * SSZ + t * 16 + colL;
      const float e = __expf(sc[idx] - m8[i]);
      sc[idx] = e;
      s8[i] += e;
    }
  }
#pragma unroll
  for (int i = 0; i < 8; ++i) {
    float sum = s8[i];
    sum += __shfl_xor(sum, 1, 16);
    sum += __shfl_xor(sum, 2, 16);
    sum += __shfl_xor(sum, 4, 16);
    sum += __shfl_xor(sum, 8, 16);
    s8[i] = 1.0f / sum;
  }

  // Pass 3: normalize, write A (single touch of the 537 MB output), keep P in LDS
  for (int t = 0; t < SSZ / 16; ++t) {
#pragma unroll
    for (int i = 0; i < 8; ++i) {
      const int idx = (rsel + i) * SSZ + t * 16 + colL;
      const float p = sc[idx] * s8[i];
      sc[idx] = p;
      Aout[(size_t)(bh * LL + q0 + rsel + i) * SSZ + t * 16 + colL] = p;
    }
  }

  // O = P @ V  (P fragments from LDS in A-layout, V from transposed buffer)
  v8f co[4];
#pragma unroll
  for (int nt = 0; nt < 4; ++nt) co[nt] = zero_v8f();

  for (int kk = 0; kk < SSZ; kk += 32) {
    v16bf pa = load_a_f32(sc + kk, SSZ, lane);
#pragma unroll
    for (int nt = 0; nt < 4; ++nt) {
      v16bf bv = load_b_bf16(Vt + ((size_t)bh * DH + nt * 16) * SSZ + kk, SSZ, lane);
      co[nt] = wmma_bf16(pa, bv, co[nt]);
    }
  }

#pragma unroll
  for (int nt = 0; nt < 4; ++nt) {
    const int col = h * DH + nt * 16 + colL;
#pragma unroll
    for (int i = 0; i < 8; ++i) {
      const int r = q0 + rsel + i;
      Of[((size_t)(b_ * LL + r)) * DM + col] = co[nt][i];
    }
  }
}

// ---------------- Host-side orchestration ----------------
extern "C" void kernel_launch(void* const* d_in, const int* in_sizes, int n_in,
                              void* d_out, int out_size, void* d_ws, size_t ws_size,
                              hipStream_t stream) {
  (void)in_sizes; (void)n_in; (void)out_size; (void)ws_size;
  const float* queries = (const float*)d_in[0];
  const float* keys    = (const float*)d_in[1];
  const float* values  = (const float*)d_in[2];
  const float* Wq = (const float*)d_in[3];
  const float* bq = (const float*)d_in[4];
  const float* Wk = (const float*)d_in[5];
  const float* bk = (const float*)d_in[6];
  const float* Wv = (const float*)d_in[7];
  const float* bv = (const float*)d_in[8];
  const float* Wo = (const float*)d_in[9];
  const float* bo = (const float*)d_in[10];

  char* ws = (char*)d_ws;
  size_t off = 0;
  bf16s* Wtq = (bf16s*)(ws + off); off += (size_t)DM * DM * sizeof(bf16s);
  bf16s* Wtk = (bf16s*)(ws + off); off += (size_t)DM * DM * sizeof(bf16s);
  bf16s* Wtv = (bf16s*)(ws + off); off += (size_t)DM * DM * sizeof(bf16s);
  bf16s* Wto = (bf16s*)(ws + off); off += (size_t)DM * DM * sizeof(bf16s);
  bf16s* QbB = (bf16s*)(ws + off); off += (size_t)BB * HH * LL * DH * sizeof(bf16s);
  bf16s* KbB = (bf16s*)(ws + off); off += (size_t)BB * HH * SSZ * DH * sizeof(bf16s);
  bf16s* VtB = (bf16s*)(ws + off); off += (size_t)BB * HH * DH * SSZ * sizeof(bf16s);
  float* Of  = (float*)(ws + off); off += (size_t)BB * LL * DM * sizeof(float);

  float* out  = (float*)d_out;
  float* Aout = out + (size_t)BB * LL * DM;

  // 0) transpose + bf16-convert weights
  wtrans_kernel<<<(DM * DM) / 256, 256, 0, stream>>>(Wq, Wtq);
  wtrans_kernel<<<(DM * DM) / 256, 256, 0, stream>>>(Wk, Wtk);
  wtrans_kernel<<<(DM * DM) / 256, 256, 0, stream>>>(Wv, Wtv);
  wtrans_kernel<<<(DM * DM) / 256, 256, 0, stream>>>(Wo, Wto);

  // 1) QKV projections
  dim3 gg((BB * LL) / 64, DM / 128);
  gemm_proj<1><<<gg, 256, 0, stream>>>(queries, Wtq, bq, nullptr, QbB);
  gemm_proj<1><<<gg, 256, 0, stream>>>(keys,    Wtk, bk, nullptr, KbB);
  gemm_proj<2><<<gg, 256, 0, stream>>>(values,  Wtv, bv, nullptr, VtB);

  // 2) fused attention; 128 KB scores + 32 KB TDM staging = 160 KB dynamic LDS
  attn_fused<<<BB * HH * (LL / 16), 32,
               SC_FLOATS * sizeof(float) + 2 * STAGE_BYTES, stream>>>(
      QbB, KbB, VtB, Aout, Of);

  // 3) output projection -> d_out
  gemm_proj<0><<<gg, 256, 0, stream>>>(Of, Wto, bo, out, nullptr);
}
